// WTConv_4561255268626
// MI455X (gfx1250) — matrix operand order repaired and checked by reading, not compiled, not run on hardware
//
#include <hip/hip_runtime.h>

typedef __attribute__((ext_vector_type(16))) _Float16 v16h;
typedef __attribute__((ext_vector_type(8)))  float    v8f;

#define B_    8
#define C_    64
#define H_    128
#define W_    128
#define SH    64       // subband spatial
#define PW    66       // padded subband width/height (1-px zero halo)
#define KT_   18       // K tiles: 576 / 32
#define CP    72       // LDS channel-padded pixel stride (halves): breaks bank stride, 16B aligned

// ---------------------------------------------------------------------------
// Kernel 1: Haar DWT -> channel-last f16 padded subbands + per-channel sums.
// grid = B*64*2 blocks (b, subband-row y, half), 256 threads.
// LDS transpose: load x rows coalesced, emit feat_pad[bs][py][px][c] coalesced.
// ---------------------------------------------------------------------------
__global__ __launch_bounds__(256) void k_dwt(const float* __restrict__ x,
                                             _Float16* __restrict__ feat_pad,
                                             float* __restrict__ chansum) {
  __shared__ float lds[64 * 2 * 65];      // [c][row 0/1][64 cols + 1 pad]
  __shared__ float red[4][256];
  const int b    = blockIdx.x >> 7;
  const int rem  = blockIdx.x & 127;
  const int y    = rem >> 1;
  const int half = rem & 1;
  const int t    = threadIdx.x;

  // Phase 1: load 64ch x 2rows x 64cols of x (fp32), coalesced float4s.
  const float* xb = x + (size_t)b * C_ * H_ * W_;
  #pragma unroll
  for (int i = 0; i < 8; ++i) {
    int id  = t + i * 256;                // 0..2047 float4 ids
    int c   = id >> 5;                    // /32
    int r   = (id >> 4) & 1;
    int seg = id & 15;
    const float4 v = *reinterpret_cast<const float4*>(
        xb + ((size_t)c * H_ + (2 * y + r)) * W_ + half * 64 + seg * 4);
    float* l = &lds[(c * 2 + r) * 65 + seg * 4];
    l[0] = v.x; l[1] = v.y; l[2] = v.z; l[3] = v.w;
  }
  __syncthreads();

  // Phase 2: per-channel Haar butterflies; channel-last f16 stores.
  const int c = t & 63;
  const int g = t >> 6;                   // 0..3, 8 pixels each
  float sll = 0.f, s0 = 0.f, s1 = 0.f, shh = 0.f;
  const size_t sstride = (size_t)PW * PW * 64;
  #pragma unroll
  for (int i = 0; i < 8; ++i) {
    int pxl = g * 8 + i;                  // local col 0..31
    int pxg = half * 32 + pxl;            // global subband col 0..63
    float a  = lds[(c * 2 + 0) * 65 + 2 * pxl];
    float bb = lds[(c * 2 + 0) * 65 + 2 * pxl + 1];
    float cc = lds[(c * 2 + 1) * 65 + 2 * pxl];
    float d  = lds[(c * 2 + 1) * 65 + 2 * pxl + 1];
    float ll = (a + bb + cc + d) * 0.5f;
    float h0 = (a + bb - cc - d) * 0.5f;
    float h1 = (a - bb + cc - d) * 0.5f;
    float hh = (a - bb - cc + d) * 0.5f;
    sll += ll; s0 += h0; s1 += h1; shh += hh;
    size_t base = (((size_t)(b * 4) * PW + (y + 1)) * PW + (pxg + 1)) * 64 + c;
    feat_pad[base + 0 * sstride] = (_Float16)ll;
    feat_pad[base + 1 * sstride] = (_Float16)h0;
    feat_pad[base + 2 * sstride] = (_Float16)h1;
    feat_pad[base + 3 * sstride] = (_Float16)hh;
  }
  red[0][t] = sll; red[1][t] = s0; red[2][t] = s1; red[3][t] = shh;
  __syncthreads();
  if (t < 64) {
    #pragma unroll
    for (int s = 0; s < 4; ++s) {
      float v = red[s][t] + red[s][t + 64] + red[s][t + 128] + red[s][t + 192];
      atomicAdd(&chansum[(b * 4 + s) * 64 + t], v);
    }
  }
}

// ---------------------------------------------------------------------------
// Kernel 2: attention softmax + kernel mixing, written directly in the
// V_WMMA_F32_16X16X32_F16 A-fragment lane layout, with K' = (u*3+v)*64 + ci.
// grid = 32 (b,s) blocks, 256 threads; 144 fragment elements per thread.
// ---------------------------------------------------------------------------
__global__ __launch_bounds__(256) void k_mix(const float* __restrict__ wt,      // (4,4,64,64,3,3)
                                             const float* __restrict__ attn_w,  // (4,64)
                                             const float* __restrict__ attn_b,  // (4)
                                             const float* __restrict__ chansum, // (32,64)
                                             _Float16* __restrict__ A_frag) {
  const int bs = blockIdx.x;
  const int s  = bs & 3;
  const int t  = threadIdx.x;

  float att[4];
  {
    float logit[4];
    #pragma unroll
    for (int k = 0; k < 4; ++k) {
      float acc = 0.f;
      for (int c = 0; c < 64; ++c)
        acc += attn_w[k * 64 + c] * chansum[bs * 64 + c];
      logit[k] = acc * (1.0f / 4096.0f) + attn_b[k];
    }
    float m  = fmaxf(fmaxf(logit[0], logit[1]), fmaxf(logit[2], logit[3]));
    float e0 = __expf(logit[0] - m), e1 = __expf(logit[1] - m);
    float e2 = __expf(logit[2] - m), e3 = __expf(logit[3] - m);
    float inv = 1.0f / (e0 + e1 + e2 + e3);
    att[0] = e0 * inv; att[1] = e1 * inv; att[2] = e2 * inv; att[3] = e3 * inv;
  }

  for (int i = 0; i < 144; ++i) {
    int idx = t + i * 256;                // 0..36863
    int e   = idx & 15;
    int L   = (idx >> 4) & 31;
    int mt  = (idx >> 9) & 3;
    int kt  = idx >> 11;                  // 0..17
    int m   = L & 15;
    int kk;
    kk = (L & 16) + e;                                // K 16*h + e (contiguous)
    int kp = kt * 32 + kk;
    int uv = kp >> 6;
    int ci = kp & 63;
    int u  = uv / 3, v = uv % 3;
    int co = mt * 16 + m;
    float acc = 0.f;
    #pragma unroll
    for (int k = 0; k < 4; ++k)
      acc += att[k] * wt[((((k * 4 + s) * 64 + co) * 64 + ci) * 3 + u) * 3 + v];
    A_frag[((((size_t)bs * KT_ + kt) * 4 + mt) * 32 + L) * 16 + e] = (_Float16)acc;
  }
}

// ---------------------------------------------------------------------------
// Kernel 3: implicit-GEMM grouped conv via WMMA f16->f32 with LDS-staged B.
// grid = (32 bs, 64 subband rows). Block = 256 threads = 8 waves.
// Block computes 64 couts x 64 pixels (one full subband row).
// The entire K-loop's feature data = 3 padded rows x 66 cols x 64 ch (~25KB),
// staged once into LDS (channel-last, pixel stride padded to 72 halves).
// Each wave: 16 couts x 32 pixels => per k-step: 1 A global v16h load,
// 2 LDS B fragments (ds_load_b128 pairs), 2 WMMAs (A register reuse).
// ---------------------------------------------------------------------------
__global__ __launch_bounds__(256) void k_gemm(const _Float16* __restrict__ A_frag,
                                              const _Float16* __restrict__ feat_pad,
                                              float* __restrict__ out_sb) {
  __shared__ _Float16 lt[3 * PW * CP];    // [u 0..2][padded col 0..65][CP halves]
  const int bs = blockIdx.x;
  const int y  = blockIdx.y;              // subband row 0..63
  const int t  = threadIdx.x;
  const int lane = t & 31;
  const int wave = t >> 5;
  const int mt = wave & 3;                // cout tile
  const int nt = wave >> 2;               // pixel-half tile (0/1)

  const _Float16* fb = feat_pad + (size_t)bs * PW * PW * 64;

  // Stage 3 padded rows x 66 cols x 64 ch into LDS (coalesced uint4 copies).
  for (int i = 0; i < 7; ++i) {
    int id = t + i * 256;                 // 16B-chunk id
    if (id < 3 * PW * 8) {
      int row = id / (PW * 8);
      int rem = id - row * (PW * 8);
      int pix = rem >> 3;
      int ch  = rem & 7;                  // 8-half chunk within pixel
      uint4 v = *reinterpret_cast<const uint4*>(
          fb + (((size_t)(y + row) * PW) + pix) * 64 + ch * 8);
      *reinterpret_cast<uint4*>(&lt[(row * PW + pix) * CP + ch * 8]) = v;
    }
  }
  __syncthreads();

  const int xq = lane & 15;               // pixel col within fragment
  const int cshift = lane & 16;           // B-frag K half select
  const _Float16* ab = A_frag + (size_t)bs * KT_ * 2048 + (size_t)mt * 512
                              + (size_t)lane * 16;

  union F16V { v16h v; uint4 q[2]; };
  v8f acc0 = {}, acc1 = {};
  #pragma unroll
  for (int kt = 0; kt < KT_; ++kt) {
    const int uvb = kt >> 1;
    const int u   = uvb / 3;
    const int v   = uvb % 3;
    const int ci0 = (kt & 1) * 32 + cshift;
    v16h a = *reinterpret_cast<const v16h*>(ab + (size_t)kt * 2048);
    const _Float16* l0 = &lt[(u * PW + (nt * 32 + xq + v)) * CP + ci0];
    const _Float16* l1 = &lt[(u * PW + (nt * 32 + 16 + xq + v)) * CP + ci0];
    F16V b0, b1;
    b0.q[0] = *reinterpret_cast<const uint4*>(l0);
    b0.q[1] = *reinterpret_cast<const uint4*>(l0 + 8);
    b1.q[0] = *reinterpret_cast<const uint4*>(l1);
    b1.q[1] = *reinterpret_cast<const uint4*>(l1 + 8);
    acc0 = __builtin_amdgcn_wmma_f32_16x16x32_f16(false, a, false, b0.v,
                                                  (short)0, acc0, false, false);
    acc1 = __builtin_amdgcn_wmma_f32_16x16x32_f16(false, a, false, b1.v,
                                                  (short)0, acc1, false, false);
  }

  const int co_hi = (lane >> 4) << 3;     // 0 or 8
  float* ob = out_sb + (size_t)bs * 64 * 4096 + (size_t)y * 64;
  #pragma unroll
  for (int g = 0; g < 8; ++g) {
    int co = mt * 16 + g + co_hi;
    int p0 = nt * 32 + xq;
    ob[(size_t)co * 4096 + p0]      = acc0[g];
    ob[(size_t)co * 4096 + p0 + 16] = acc1[g];
  }
}

// ---------------------------------------------------------------------------
// Kernel 4: Haar IDWT, coalesced float2 stores of interleaved output.
// ---------------------------------------------------------------------------
__global__ __launch_bounds__(256) void k_idwt(const float* __restrict__ out_sb,
                                              float* __restrict__ out) {
  int idx = blockIdx.x * 256 + threadIdx.x;  // 2,097,152 total
  int p   = idx & 4095;
  int bc  = idx >> 12;
  int b   = bc >> 6;
  int c   = bc & 63;
  int y   = p >> 6;
  int x   = p & 63;
  size_t base = (((size_t)(b * 4)) * 64 + c) * 4096 + p;
  const size_t ss = (size_t)64 * 4096;
  float ll = out_sb[base];
  float h0 = out_sb[base + ss];
  float h1 = out_sb[base + 2 * ss];
  float hh = out_sb[base + 3 * ss];
  float a  = (ll + h0 + h1 + hh) * 0.5f;
  float bb = (ll + h0 - h1 - hh) * 0.5f;
  float cc = (ll - h0 + h1 - hh) * 0.5f;
  float d  = (ll - h0 - h1 + hh) * 0.5f;
  float* orow = out + (((size_t)(b * 64 + c) * 128) + 2 * y) * 128;
  *reinterpret_cast<float2*>(orow + 2 * x)       = make_float2(a, bb);
  *reinterpret_cast<float2*>(orow + 128 + 2 * x) = make_float2(cc, d);
}

// ---------------------------------------------------------------------------
extern "C" void kernel_launch(void* const* d_in, const int* in_sizes, int n_in,
                              void* d_out, int out_size, void* d_ws, size_t ws_size,
                              hipStream_t stream) {
  const float* x   = (const float*)d_in[0];   // (8,64,128,128)
  const float* wt  = (const float*)d_in[1];   // (4,4,64,64,3,3)
  const float* aw  = (const float*)d_in[2];   // (4,64)
  const float* abv = (const float*)d_in[3];   // (4,)

  const size_t FEAT_BYTES = (size_t)32 * PW * PW * 64 * sizeof(_Float16); // 17,842,176
  const size_t AFRG_BYTES = (size_t)32 * KT_ * 4 * 32 * 16 * sizeof(_Float16); // 2,359,296
  const size_t CSUM_BYTES = (size_t)32 * 64 * sizeof(float);              // 8,192

  char* ws = (char*)d_ws;
  _Float16* feat_pad = (_Float16*)ws;
  _Float16* A_frag   = (_Float16*)(ws + FEAT_BYTES);
  float*    chansum  = (float*)(ws + FEAT_BYTES + AFRG_BYTES);
  float*    out_sb   = (float*)(ws + FEAT_BYTES + AFRG_BYTES + CSUM_BYTES);

  hipMemsetAsync(feat_pad, 0, FEAT_BYTES, stream);  // zero halo for padding
  hipMemsetAsync(chansum, 0, CSUM_BYTES, stream);

  k_dwt <<<dim3(B_ * 64 * 2), 256, 0, stream>>>(x, feat_pad, chansum);
  k_mix <<<dim3(32),          256, 0, stream>>>(wt, aw, abv, chansum, A_frag);
  k_gemm<<<dim3(32, 64),      256, 0, stream>>>(A_frag, feat_pad, out_sb);
  k_idwt<<<dim3(8192),        256, 0, stream>>>(out_sb, (float*)d_out);
}
